// SelfAttention_11364483465551
// MI455X (gfx1250) — matrix operand order repaired and checked by reading, not compiled, not run on hardware
//
#include <hip/hip_runtime.h>
#include <hip/hip_bf16.h>
#include <cstdint>
#include <cstddef>

// ---------------------------------------------------------------------------
// Problem constants (from reference)
// ---------------------------------------------------------------------------
#define BB     2
#define SS     2048          // S = F*H*W = 8*16*16
#define DIMN   2048
#define NHN    16
#define HDN    128
#define CN     64
#define CTN    22
#define CHN    21
#define SHIFTN 4
#define MM     (BB * SS)     // 4096 rows in all GEMMs

typedef __attribute__((ext_vector_type(16))) __bf16 v16bf;
typedef __attribute__((ext_vector_type(8)))  __bf16 v8bf;
typedef __attribute__((ext_vector_type(8)))  float  v8f;
typedef __attribute__((ext_vector_type(4)))  unsigned int u32x4;
typedef __attribute__((ext_vector_type(8)))  int i32x8;
typedef __attribute__((ext_vector_type(4)))  int i32x4;

#if __has_builtin(__builtin_amdgcn_tensor_load_to_lds) && \
    __has_builtin(__builtin_amdgcn_s_wait_tensorcnt)
#define USE_TDM 1
#else
#define USE_TDM 0
#endif

__device__ __forceinline__ v16bf cat8(v8bf lo, v8bf hi) {
  v16bf r;
#pragma unroll
  for (int i = 0; i < 8; ++i) { r[i] = lo[i]; r[i + 8] = hi[i]; }
  return r;
}

__device__ __forceinline__ v8f wmma_bf16(v16bf a, v16bf b, v8f c) {
  // D(16x16 f32) = A(16x32 bf16) * B(32x16 bf16) + C
  return __builtin_amdgcn_wmma_f32_16x16x32_bf16(false, a, false, b,
                                                 (short)0, c, false, false);
}

__device__ __forceinline__ v8f zero8() {
  v8f z;
#pragma unroll
  for (int i = 0; i < 8; ++i) z[i] = 0.0f;
  return z;
}

#if USE_TDM
// ---------------------------------------------------------------------------
// TDM: 2-D tile load Wt[row0 .. row0+127][k .. k+63 halves] -> LDS
// D# per CDNA5 ISA. data_size = 4 bytes.
//   tile_dim0 = 32 dwords (64 bf16), tile_dim1 = 128 rows
//   tensor_dim0 = 1024 dwords (full Wt row), tensor_dim1 = 2048 rows
//   tensor_dim0_stride = 1024 dwords (Wt row pitch)
// 6-arg builtin form (clang-23 / therock-10.0 headers).
// ---------------------------------------------------------------------------
__device__ __forceinline__ void tdm_load_tile(unsigned lds_addr,
                                              const __bf16* gptr) {
  const uint64_t ga = (uint64_t)(uintptr_t)gptr;
  u32x4 g0 = {0u, 0u, 0u, 0u};
  g0[0] = 1u;                                              // count=1 (valid)
  g0[1] = lds_addr;                                        // LDS byte addr
  g0[2] = (unsigned)ga;                                    // global addr lo
  g0[3] = ((unsigned)(ga >> 32) & 0x1FFFFFFu) | (2u << 30);  // hi | type=2
  i32x8 g1 = {0, 0, 0, 0, 0, 0, 0, 0};
  g1[0] = (2 << 16);                       // wg_mask=0, data_size=2 (4B)
  g1[1] = (int)((1024u & 0xFFFFu) << 16);  // tensor_dim0[15:0] = 1024
  g1[2] = (int)((1024u >> 16) | ((2048u & 0xFFFFu) << 16));  // d0 hi | d1 lo
  g1[3] = (int)((2048u >> 16) | (32u << 16));  // d1 hi | tile_dim0 = 32 dw
  g1[4] = (int)128u;                       // tile_dim1 = 128, tile_dim2 = 0
  g1[5] = 1024;                            // tensor_dim0_stride = 1024 dw
  g1[6] = 0;                               // stride hi, dim1_stride lo
  g1[7] = 0;
  i32x4 g2 = {0, 0, 0, 0};
  i32x4 g3 = {0, 0, 0, 0};
  i32x8 g4 = {0, 0, 0, 0, 0, 0, 0, 0};
  __builtin_amdgcn_tensor_load_to_lds(g0, g1, g2, g3, g4, 0);
}
#endif

// ---------------------------------------------------------------------------
// 1) elementwise f32 -> bf16 cast
// ---------------------------------------------------------------------------
__global__ __launch_bounds__(256) void cast_bf16_kernel(
    const float* __restrict__ in, __bf16* __restrict__ out, int n) {
  int i = blockIdx.x * 256 + threadIdx.x;
  if (i < n) out[i] = (__bf16)in[i];
}

// ---------------------------------------------------------------------------
// 2) weight transpose + cast:  Wt[n][k] = (bf16) W[k][n]   (2048 x 2048)
// ---------------------------------------------------------------------------
__global__ __launch_bounds__(256) void transpose_cast_kernel(
    const float* __restrict__ W, __bf16* __restrict__ Wt) {
  __shared__ __bf16 tile[32][33];
  const int bx = blockIdx.x * 32;   // n-tile
  const int by = blockIdx.y * 32;   // k-tile
  const int tx = threadIdx.x & 31;
  const int ty = threadIdx.x >> 5;  // 0..7
#pragma unroll
  for (int i = 0; i < 4; ++i) {
    int r = ty + i * 8;
    tile[r][tx] = (__bf16)W[(size_t)(by + r) * DIMN + bx + tx];
  }
  __syncthreads();
#pragma unroll
  for (int i = 0; i < 4; ++i) {
    int r = ty + i * 8;
    Wt[(size_t)(bx + r) * DIMN + by + tx] = tile[tx][r];
  }
}

// ---------------------------------------------------------------------------
// 3) GEMM + bias:  Y[m][n] = sum_k A[m][k] * Wt[n][k] + bias[n]
//    Block tile 128(M) x 128(N), 8 waves; wave owns 16 M-rows x 128 N.
//    B tile (128 rows x 64 halves) staged in LDS, double buffered:
//      wave 0 issues TENSOR_LOAD_TO_LDS for tile i+1, s_wait_tensorcnt(1)
//      guarantees tile i landed (TDM completes in order) while i+1 streams.
//    16 v_wmma per K-stage per wave.
// ---------------------------------------------------------------------------
__global__ __launch_bounds__(256) void gemm_bias_kernel(
    const __bf16* __restrict__ A, const __bf16* __restrict__ Wt,
    const float* __restrict__ bias, float* __restrict__ Y) {
  __shared__ __bf16 ldsB[2][128 * 64];  // 2 x 16 KB, row-major tiles

  const int t    = threadIdx.x;
  const int wave = t >> 5;
  const int lane = t & 31;
  const int bm   = blockIdx.y * 128;
  const int bn   = blockIdx.x * 128;

  const int mrow = lane & 15;   // A-fragment row / C-fragment column
  const int asel = lane >> 4;   // K-half-group selector

  const __bf16* arow = A + (size_t)(bm + wave * 16 + mrow) * DIMN + asel * 16;
  const __bf16* wbase = Wt + (size_t)bn * DIMN;

#if USE_TDM
  const unsigned lds_base = (unsigned)(uintptr_t)(&ldsB[0][0]);
#endif

  v8f acc[8];
#pragma unroll
  for (int s = 0; s < 8; ++s) acc[s] = zero8();

#if USE_TDM
  if (wave == 0) tdm_load_tile(lds_base, wbase);  // tile 0 -> buf 0
#endif

  const int ntiles = DIMN / 64;  // 32
  for (int it = 0; it < ntiles; ++it) {
    const int k0  = it * 64;
    const int buf = it & 1;
    __syncthreads();  // previous stage's reads complete before overwrite
#if USE_TDM
    if (wave == 0) {
      if (it + 1 < ntiles) {
        tdm_load_tile(lds_base + (unsigned)((buf ^ 1) * 128 * 64 * 2),
                      wbase + (size_t)(k0 + 64));
        __builtin_amdgcn_s_wait_tensorcnt(1);  // tile `it` resident
      } else {
        __builtin_amdgcn_s_wait_tensorcnt(0);
      }
    }
#else
    {  // cooperative staging fallback: thread t copies 32B of one row
      const int r  = t >> 1;
      const int c  = (t & 1) * 32;  // halves
      const __bf16* src = wbase + (size_t)r * DIMN + k0 + c;
      __bf16* dst = &ldsB[buf][r * 64 + c];
      *(v16bf*)dst = *(const v16bf*)src;
    }
#endif
    __syncthreads();

    // pull next A stage into WGP-level cache (locality 3 -> WGP scope)
    __builtin_prefetch(arow + k0 + 64, 0, 3);

#pragma unroll
    for (int kk = 0; kk < 64; kk += 32) {
      v8bf alo = *(const v8bf*)(arow + k0 + kk);
      v8bf ahi = *(const v8bf*)(arow + k0 + kk + 8);
      v16bf afrag = cat8(alo, ahi);
#pragma unroll
      for (int s = 0; s < 8; ++s) {
        v16bf bfrag =
            *(const v16bf*)&ldsB[buf][(s * 16 + mrow) * 64 + kk + asel * 16];
        acc[s] = wmma_bf16(afrag, bfrag, acc[s]);
      }
    }
  }

  // epilogue: C layout lane l: col = l%16, rows r + 8*(l>>4)
#pragma unroll
  for (int s = 0; s < 8; ++s) {
    const int col = bn + s * 16 + mrow;
    const float bv = bias[col];
#pragma unroll
    for (int r = 0; r < 8; ++r) {
      const int m = bm + wave * 16 + r + 8 * asel;
      Y[(size_t)m * DIMN + col] = acc[s][r] + bv;
    }
  }
}

// ---------------------------------------------------------------------------
// 4) fused rmsnorm + RoPE, relayout to head-major bf16 [B][NH][S][HD]
// ---------------------------------------------------------------------------
__global__ __launch_bounds__(256) void norm_rope_kernel(
    const float* __restrict__ Y, const float* __restrict__ nw,
    const float* __restrict__ freqs, const int* __restrict__ inner_t,
    const int* __restrict__ seq_lens, __bf16* __restrict__ out) {
  __shared__ float red[256];
  const int bs = blockIdx.x;
  const int b  = bs / SS;
  const int s  = bs % SS;
  const int t  = threadIdx.x;
  const float* row = Y + (size_t)bs * DIMN;

  float ss = 0.f;
  const int base = t * 8;
#pragma unroll
  for (int i = 0; i < 8; ++i) { float v = row[base + i]; ss += v * v; }
  red[t] = ss;
  __syncthreads();
  for (int off = 128; off > 0; off >>= 1) {
    if (t < off) red[t] += red[t + off];
    __syncthreads();
  }
  const float inv = rsqrtf(red[0] * (1.0f / DIMN) + 1e-6f);

  const int fI = s >> 8;            // frame
  const int h  = (s >> 4) & 15;
  const int w  = s & 15;
  const int c0 = inner_t[b * 2];    // first cumsum entry
  const int tshot = (fI >= c0) ? 1 : 0;
  const int tidx  = tshot * SHIFTN + fI;
  const bool keep = (s < seq_lens[b]);

#pragma unroll
  for (int p = 0; p < 4; ++p) {
    const int d  = base + p * 2;
    const int n  = d >> 7;          // head
    const int hd = d & 127;
    const int c  = hd >> 1;         // complex channel 0..63
    const int pos = (c < CTN) ? tidx : ((c < CTN + CHN) ? h : w);
    const float cr = freqs[((size_t)pos * CN + c) * 2 + 0];
    const float ci = freqs[((size_t)pos * CN + c) * 2 + 1];
    const float a  = row[d]     * inv * nw[d];
    const float bq = row[d + 1] * inv * nw[d + 1];
    const float ra = keep ? (a * cr - bq * ci) : a;
    const float rb = keep ? (a * ci + bq * cr) : bq;
    __bf16* o = out + ((size_t)(b * NHN + n) * SS + s) * HDN + hd;
    o[0] = (__bf16)ra;
    o[1] = (__bf16)rb;
  }
}

// ---------------------------------------------------------------------------
// 5) V relayout: f32 [B*S][DIM] -> bf16 transposed [B][NH][HD][S]
// ---------------------------------------------------------------------------
__global__ __launch_bounds__(256) void v_relayout_kernel(
    const float* __restrict__ V, __bf16* __restrict__ Vt) {
  const int idx = blockIdx.x * 256 + threadIdx.x;
  const int bs = idx >> 11;
  const int d  = idx & 2047;
  const int b  = bs / SS;
  const int s  = bs % SS;
  const int n  = d >> 7;
  const int hd = d & 127;
  Vt[((size_t)(b * NHN + n) * HDN + hd) * SS + s] = (__bf16)V[idx];
}

// ---------------------------------------------------------------------------
// 6) flash attention (bf16 WMMA, f32 online softmax)
// ---------------------------------------------------------------------------
__global__ __launch_bounds__(256) void attn_kernel(
    const __bf16* __restrict__ Qh, const __bf16* __restrict__ Kh,
    const __bf16* __restrict__ Vt, const int* __restrict__ seq_lens,
    __bf16* __restrict__ Out) {
  __shared__ __bf16 ldsP[8][16 * 32];  // per-wave P transpose staging (8 KB)

  const int t    = threadIdx.x;
  const int wave = t >> 5;
  const int lane = t & 31;
  const int blk  = blockIdx.x;
  const int qt   = blk & 15;
  const int head = (blk >> 4) & 15;
  const int b    = blk >> 8;

  const int q0   = qt * 128 + wave * 16;
  const int mrow = lane & 15;
  const int asel = lane >> 4;

  const __bf16* Qbase = Qh + (size_t)(b * NHN + head) * SS * HDN;
  const __bf16* Kbase = Kh + (size_t)(b * NHN + head) * SS * HDN;
  const __bf16* Vbase = Vt + (size_t)(b * NHN + head) * HDN * SS;

  v16bf qf[4];
  const __bf16* qrow = Qbase + (size_t)(q0 + mrow) * HDN;
#pragma unroll
  for (int kk = 0; kk < 4; ++kk) {
    v8bf lo = *(const v8bf*)(qrow + kk * 32 + asel * 16);
    v8bf hi = *(const v8bf*)(qrow + kk * 32 + asel * 16 + 8);
    qf[kk] = cat8(lo, hi);
  }

  v8f acc[8];
  float m[8], l[8];
#pragma unroll
  for (int s = 0; s < 8; ++s) acc[s] = zero8();
#pragma unroll
  for (int r = 0; r < 8; ++r) { m[r] = -1e30f; l[r] = 0.f; }

  const float scale = 0.088388347648318447f;  // 128^-0.5
  const int slen = seq_lens[b];
  __bf16* pw = &ldsP[wave][0];

  for (int kt = 0; kt < SS; kt += 32) {
    // scores: two 16x16 subtiles, 4 WMMAs each over HD
    v8f sc[2];
#pragma unroll
    for (int ns = 0; ns < 2; ++ns) {
      v8f s8 = zero8();
      const __bf16* krow =
          Kbase + (size_t)(kt + ns * 16 + mrow) * HDN + asel * 16;
#pragma unroll
      for (int kk = 0; kk < 4; ++kk) {
        v16bf bfrag = *(const v16bf*)(krow + kk * 32);
        s8 = wmma_bf16(qf[kk], bfrag, s8);
      }
      sc[ns] = s8;
    }

    // online softmax (16-lane row groups)
    const int col0 = kt + mrow;
    const int col1 = kt + 16 + mrow;
    float corr[8];
#pragma unroll
    for (int r = 0; r < 8; ++r) {
      float v0 = sc[0][r] * scale;
      float v1 = sc[1][r] * scale;
      if (col0 >= slen) v0 = -1e30f;
      if (col1 >= slen) v1 = -1e30f;
      float mx = fmaxf(v0, v1);
      mx = fmaxf(mx, __shfl_xor(mx, 1, 32));
      mx = fmaxf(mx, __shfl_xor(mx, 2, 32));
      mx = fmaxf(mx, __shfl_xor(mx, 4, 32));
      mx = fmaxf(mx, __shfl_xor(mx, 8, 32));
      const float mn = fmaxf(m[r], mx);
      const float p0 = __expf(v0 - mn);
      const float p1 = __expf(v1 - mn);
      float rs = p0 + p1;
      rs += __shfl_xor(rs, 1, 32);
      rs += __shfl_xor(rs, 2, 32);
      rs += __shfl_xor(rs, 4, 32);
      rs += __shfl_xor(rs, 8, 32);
      corr[r] = __expf(m[r] - mn);
      l[r] = l[r] * corr[r] + rs;
      m[r] = mn;
      sc[0][r] = p0;
      sc[1][r] = p1;
    }
#pragma unroll
    for (int s = 0; s < 8; ++s)
#pragma unroll
      for (int r = 0; r < 8; ++r) acc[s][r] *= corr[r];

    // P: C-layout -> A-layout via per-wave LDS tile
#pragma unroll
    for (int r = 0; r < 8; ++r) {
      const int prow = r + 8 * asel;
      pw[prow * 32 + mrow]      = (__bf16)sc[0][r];
      pw[prow * 32 + 16 + mrow] = (__bf16)sc[1][r];
    }
    asm volatile("s_wait_dscnt 0" ::: "memory");
    v8bf plo = *(const v8bf*)&pw[mrow * 32 + asel * 16];
    v8bf phi = *(const v8bf*)&pw[mrow * 32 + asel * 16 + 8];
    v16bf pf = cat8(plo, phi);

    // O += P(16x32) * V(32x128): 8 WMMAs over HD subtiles
#pragma unroll
    for (int s = 0; s < 8; ++s) {
      const __bf16* vrow =
          Vbase + (size_t)(s * 16 + mrow) * SS + kt + asel * 16;
      v16bf vfrag = *(const v16bf*)vrow;
      acc[s] = wmma_bf16(pf, vfrag, acc[s]);
    }
  }

  // finalize: divide by l, write bf16 [B*S][DIM]
#pragma unroll
  for (int s = 0; s < 8; ++s) {
#pragma unroll
    for (int r = 0; r < 8; ++r) {
      const int srow = q0 + r + 8 * asel;
      const int col  = head * HDN + s * 16 + mrow;
      Out[((size_t)(b * SS) + srow) * DIMN + col] = (__bf16)(acc[s][r] / l[r]);
    }
  }
}

// ---------------------------------------------------------------------------
// host-side launcher
// ---------------------------------------------------------------------------
extern "C" void kernel_launch(void* const* d_in, const int* in_sizes, int n_in,
                              void* d_out, int out_size, void* d_ws,
                              size_t ws_size, hipStream_t stream) {
  (void)in_sizes; (void)n_in; (void)out_size; (void)ws_size;

  const float* x        = (const float*)d_in[0];
  const float* q_w      = (const float*)d_in[1];
  const float* q_b      = (const float*)d_in[2];
  const float* k_w      = (const float*)d_in[3];
  const float* k_b      = (const float*)d_in[4];
  const float* v_w      = (const float*)d_in[5];
  const float* v_b      = (const float*)d_in[6];
  const float* o_w      = (const float*)d_in[7];
  const float* o_b      = (const float*)d_in[8];
  const float* norm_q_w = (const float*)d_in[9];
  const float* norm_k_w = (const float*)d_in[10];
  const float* freqs    = (const float*)d_in[11];
  const int*   seq_lens = (const int*)d_in[12];
  const int*   inner_t  = (const int*)d_in[14];

  char* w = (char*)d_ws;
  size_t off = 0;
  __bf16* XB  = (__bf16*)(w + off); off += (size_t)MM * DIMN * 2;      // 16 MB
  __bf16* WTq = (__bf16*)(w + off); off += (size_t)DIMN * DIMN * 2;    // 8 MB
  __bf16* WTk = (__bf16*)(w + off); off += (size_t)DIMN * DIMN * 2;
  __bf16* WTv = (__bf16*)(w + off); off += (size_t)DIMN * DIMN * 2;
  __bf16* WTo = (__bf16*)(w + off); off += (size_t)DIMN * DIMN * 2;
  float*  QF  = (float*)(w + off);  off += (size_t)MM * DIMN * 4;      // 32 MB
  float*  KF  = (float*)(w + off);  off += (size_t)MM * DIMN * 4;
  float*  VF  = (float*)(w + off);  off += (size_t)MM * DIMN * 4;
  __bf16* Qh  = (__bf16*)(w + off); off += (size_t)MM * DIMN * 2;
  __bf16* Kh  = (__bf16*)(w + off); off += (size_t)MM * DIMN * 2;
  __bf16* VT  = (__bf16*)(w + off); off += (size_t)MM * DIMN * 2;
  __bf16* AO  = (__bf16*)QF;  // reuse QF region for attention output (bf16)

  const int n_x = MM * DIMN;  // 8,388,608

  cast_bf16_kernel<<<n_x / 256, 256, 0, stream>>>(x, XB, n_x);
  dim3 tgrid(DIMN / 32, DIMN / 32);
  transpose_cast_kernel<<<tgrid, 256, 0, stream>>>(q_w, WTq);
  transpose_cast_kernel<<<tgrid, 256, 0, stream>>>(k_w, WTk);
  transpose_cast_kernel<<<tgrid, 256, 0, stream>>>(v_w, WTv);
  transpose_cast_kernel<<<tgrid, 256, 0, stream>>>(o_w, WTo);

  dim3 ggrid(DIMN / 128, MM / 128);
  gemm_bias_kernel<<<ggrid, 256, 0, stream>>>(XB, WTq, q_b, QF);
  gemm_bias_kernel<<<ggrid, 256, 0, stream>>>(XB, WTk, k_b, KF);
  gemm_bias_kernel<<<ggrid, 256, 0, stream>>>(XB, WTv, v_b, VF);

  norm_rope_kernel<<<BB * SS, 256, 0, stream>>>(QF, norm_q_w, freqs, inner_t,
                                                seq_lens, Qh);
  norm_rope_kernel<<<BB * SS, 256, 0, stream>>>(KF, norm_k_w, freqs, inner_t,
                                                seq_lens, Kh);
  v_relayout_kernel<<<n_x / 256, 256, 0, stream>>>(VF, VT);

  attn_kernel<<<BB * NHN * (SS / 128), 256, 0, stream>>>(Qh, Kh, VT, seq_lens,
                                                         AO);

  gemm_bias_kernel<<<ggrid, 256, 0, stream>>>(AO, WTo, o_b, (float*)d_out);
}
